// DrMVSNet_86655260164472
// MI455X (gfx1250) — matrix operand order repaired and checked by reading, not compiled, not run on hardware
//
#include <hip/hip_runtime.h>
#include <hip/hip_bf16.h>

// ---------------------------------------------------------------------------
// DrMVSNet forward for MI455X (gfx1250, wave32, WMMA).
// Convs = explicit im2col (f16) + pure WMMA GEMM:
//   D(16oc x 32px) += A(16oc x 32k) * B(32k x 32px), two accumulators/wave.
// Hot loop is b128 loads + v_wmma only; all gather math lives in im2col.
// ---------------------------------------------------------------------------

typedef __attribute__((ext_vector_type(16))) _Float16 v16h;
typedef __attribute__((ext_vector_type(8)))  _Float16 v8h;
typedef __attribute__((ext_vector_type(8)))  float    v8f;

#define H_IMG 384
#define W_IMG 512
#define HF 96
#define WF 128
#define HWF (HF*WF)
#define ND 24

static __device__ __forceinline__ float sigm_(float x){ return 1.0f/(1.0f+__expf(-x)); }

// ---------------- weight packing: f32 OIHW -> f16 [CoutPad x Kpad] ----------
__global__ void pack_w_k(const float* __restrict__ w, _Float16* __restrict__ wp,
                         int Cout, int Cin, int CoutPad, int Kpad){
  int idx = blockIdx.x*blockDim.x + threadIdx.x;
  int total = CoutPad*Kpad;
  if (idx >= total) return;
  int co = idx / Kpad, k = idx % Kpad;
  int K = Cin*9;
  float v = 0.f;
  if (co < Cout && k < K) v = w[(size_t)co*K + k];
  wp[idx] = (_Float16)v;
}

// ---------------- im2col: f32 NCHW -> f16 [NpxPad x Kpad] -------------------
__global__ void im2col_k(const float* __restrict__ in, _Float16* __restrict__ Bm,
                         int Cin, int Hin, int Win, int Hout, int Wout,
                         int stride, int Kpad, int NpxPad){
  size_t idx = (size_t)blockIdx.x*blockDim.x + threadIdx.x;
  size_t total = (size_t)NpxPad*Kpad;
  if (idx >= total) return;
  int k = (int)(idx % Kpad);
  int p = (int)(idx / Kpad);
  int K = Cin*9, Npx = Hout*Wout;
  float v = 0.f;
  if (p < Npx && k < K){
    int ci = k / 9, t = k - ci*9;
    int ky = t / 3, kx = t - ky*3;
    int y = p / Wout, x = p - y*Wout;
    int iy = y*stride + ky - 1;
    int ix = x*stride + kx - 1;
    if (iy >= 0 && iy < Hin && ix >= 0 && ix < Win)
      v = in[(size_t)ci*Hin*Win + (size_t)iy*Win + ix];
  }
  Bm[idx] = (_Float16)v;
}

// ---------------- pure WMMA GEMM: out[oc][px] = A x B^T + bias --------------
// A : [CoutPad x Kpad] f16   (row = oc, contiguous k)
// Bm: [NpxPad  x Kpad] f16   (row = pixel, contiguous k)
// Fragment layouts (wave32):
//   A 16x32: lanes 0-15 M=lane, K=0..15; lanes 16-31 same M, K=16..31.
//   B 32x16: lane N=lane&15; lanes 0-15 K=0..15, lanes 16-31 K=16..31.
//   C/D 16x16: vgpr r, lane l -> N=l&15, M=r+(l>=16?8:0).
__global__ __launch_bounds__(256)
void gemm_wmma_k(const _Float16* __restrict__ A, const _Float16* __restrict__ Bm,
                 const float* __restrict__ bias, float* __restrict__ out,
                 int Cout, int Npx, int Kpad, int nOcT, int nPxT2)
{
  int lane = threadIdx.x & 31;
  int wv   = blockIdx.x*(blockDim.x>>5) + (threadIdx.x>>5);
  if (wv >= nOcT*nPxT2) return;           // uniform per wave -> EXEC all-1 at WMMA
  int ocT = wv % nOcT;
  int pxT = wv / nOcT;

  int m   = lane & 15;
  int hiA = (lane >> 4) ? 16 : 0;
  int hiB = (lane >> 4) ? 16 : 0;
  const _Float16* arow  = A  + (size_t)(ocT*16 + m)*Kpad;
  const _Float16* brow0 = Bm + (size_t)(pxT*32 + (lane & 15))*Kpad;
  const _Float16* brow1 = brow0 + (size_t)16*Kpad;

  v8f acc0 = {}, acc1 = {};
#pragma unroll 2
  for (int k0 = 0; k0 < Kpad; k0 += 32){
    const _Float16* ap = arow + k0;
    v8h a0 = *(const v8h*)(ap + hiA);
    v8h a1 = *(const v8h*)(ap + hiA + 8);
    v16h a = __builtin_shufflevector(a0, a1, 0,1,2,3,4,5,6,7,8,9,10,11,12,13,14,15);
    v16h b0 = *(const v16h*)(brow0 + k0 + hiB);
    v16h b1 = *(const v16h*)(brow1 + k0 + hiB);
    __builtin_prefetch((const void*)(brow0 + k0 + hiB + 64), 0, 1);
    __builtin_prefetch((const void*)(brow1 + k0 + hiB + 64), 0, 1);
    acc0 = __builtin_amdgcn_wmma_f32_16x16x32_f16(false, a, false, b0,
                                                  (short)0, acc0, false, false);
    acc1 = __builtin_amdgcn_wmma_f32_16x16x32_f16(false, a, false, b1,
                                                  (short)0, acc1, false, false);
  }

  int n = lane & 15;
  int mofs = (lane >> 4) ? 8 : 0;
  int p0 = pxT*32 + n, p1 = p0 + 16;
#pragma unroll
  for (int r = 0; r < 8; ++r){
    int oc = ocT*16 + r + mofs;
    if (oc < Cout){
      float bi = bias[oc];
      if (p0 < Npx) out[(size_t)oc*Npx + p0] = acc0[r] + bi;
      if (p1 < Npx) out[(size_t)oc*Npx + p1] = acc1[r] + bi;
    }
  }
}

// ---------------- GroupNorm (groups=4) --------------------------------------
__global__ void gn_stats_k(const float* __restrict__ x, float* __restrict__ stats,
                           int C, int HW, int groups){
  int g = blockIdx.x;
  int cpg = C / groups;
  size_t n = (size_t)cpg * HW;
  const float* base = x + (size_t)g * n;
  float s = 0.f, s2 = 0.f;
  for (size_t i = threadIdx.x; i < n; i += blockDim.x){
    float v = base[i]; s += v; s2 += v*v;
  }
  __shared__ float sh[512];
  sh[threadIdx.x] = s; sh[256 + threadIdx.x] = s2;
  __syncthreads();
  for (int o = 128; o > 0; o >>= 1){
    if ((int)threadIdx.x < o){
      sh[threadIdx.x]       += sh[threadIdx.x + o];
      sh[256 + threadIdx.x] += sh[256 + threadIdx.x + o];
    }
    __syncthreads();
  }
  if (threadIdx.x == 0){
    float mean = sh[0] / (float)n;
    float var  = sh[256] / (float)n - mean*mean;
    stats[2*g] = mean; stats[2*g + 1] = rsqrtf(var + 1e-5f);
  }
}

__global__ void gn_apply_k(float* __restrict__ x, const float* __restrict__ stats,
                           const float* __restrict__ gamma, const float* __restrict__ beta,
                           int C, int HW, int groups, int relu){
  size_t idx = (size_t)blockIdx.x*blockDim.x + threadIdx.x;
  size_t total = (size_t)C*HW;
  if (idx >= total) return;
  int c = (int)(idx / HW);
  int g = c / (C / groups);
  float v = (x[idx] - stats[2*g]) * stats[2*g+1] * gamma[c] + beta[c];
  if (relu) v = v > 0.f ? v : 0.f;
  x[idx] = v;
}

// ---------------- elementwise helpers ---------------------------------------
__global__ void act_k(float* __restrict__ x, size_t n, int mode){ // 1=relu 2=sigmoid
  size_t i = (size_t)blockIdx.x*blockDim.x + threadIdx.x;
  if (i >= n) return;
  float v = x[i];
  if (mode == 1) v = v > 0.f ? v : 0.f;
  else if (mode == 2) v = sigm_(v);
  x[i] = v;
}
__global__ void copy_k(float* __restrict__ d, const float* __restrict__ s, size_t n){
  size_t i = (size_t)blockIdx.x*blockDim.x + threadIdx.x;
  if (i < n) d[i] = s[i];
}
__global__ void zero_k(float* __restrict__ d, size_t n){
  size_t i = (size_t)blockIdx.x*blockDim.x + threadIdx.x;
  if (i < n) d[i] = 0.f;
}
__global__ void neg_half_k(float* __restrict__ x, size_t n){
  size_t i = (size_t)blockIdx.x*blockDim.x + threadIdx.x;
  if (i < n) x[i] = -0.5f * x[i];
}
__global__ void acc_term_k(float* __restrict__ acc, const float* __restrict__ diff,
                           const float* __restrict__ rw, int C, int HW, int first){
  size_t idx = (size_t)blockIdx.x*blockDim.x + threadIdx.x;
  size_t total = (size_t)C*HW;
  if (idx >= total) return;
  int p = (int)(idx % HW);
  float t = (rw[p] + 1.f) * diff[idx];
  acc[idx] = first ? t : acc[idx] + t;
}
__global__ void maxpool2_k(const float* __restrict__ in, float* __restrict__ out,
                           int C, int H, int W){
  int Ho = H >> 1, Wo = W >> 1;
  size_t idx = (size_t)blockIdx.x*blockDim.x + threadIdx.x;
  size_t total = (size_t)C*Ho*Wo;
  if (idx >= total) return;
  int c = (int)(idx / (Ho*Wo));
  int r = (int)(idx - (size_t)c*Ho*Wo);
  int y = r / Wo, x = r - y*Wo;
  const float* p = in + (size_t)c*H*W;
  float a = p[(2*y)*W + 2*x],     b = p[(2*y)*W + 2*x + 1];
  float cc = p[(2*y+1)*W + 2*x],  d = p[(2*y+1)*W + 2*x + 1];
  out[idx] = fmaxf(fmaxf(a,b), fmaxf(cc,d));
}
__global__ void up2_k(const float* __restrict__ in, float* __restrict__ out,
                      int C, int Ho, int Wo){ // out is Ho x Wo (2x input)
  size_t idx = (size_t)blockIdx.x*blockDim.x + threadIdx.x;
  size_t total = (size_t)C*Ho*Wo;
  if (idx >= total) return;
  int c = (int)(idx / (Ho*Wo));
  int r = (int)(idx - (size_t)c*Ho*Wo);
  int y = r / Wo, x = r - y*Wo;
  int Hi = Ho >> 1, Wi = Wo >> 1;
  out[idx] = in[(size_t)c*Hi*Wi + (size_t)(y>>1)*Wi + (x>>1)];
}
__global__ void lstm_gates_k(const float* __restrict__ z, float* __restrict__ h,
                             float* __restrict__ c, int chid, int HW){
  size_t idx = (size_t)blockIdx.x*blockDim.x + threadIdx.x;
  size_t total = (size_t)chid*HW;
  if (idx >= total) return;
  int ch = (int)(idx / HW);
  int p  = (int)(idx - (size_t)ch*HW);
  float ig = sigm_(z[(size_t)ch*HW + p]);
  float fg = sigm_(z[(size_t)(chid + ch)*HW + p]);
  float og = sigm_(z[(size_t)(2*chid + ch)*HW + p]);
  float gg = tanhf (z[(size_t)(3*chid + ch)*HW + p]);
  float c2 = fg * c[idx] + ig * gg;
  c[idx] = c2;
  h[idx] = og * tanhf(c2);
}

// ---------------- direct small conv (Cout==1) -------------------------------
__global__ void conv3x3_1out_k(const float* __restrict__ in, const float* __restrict__ w,
                               const float* __restrict__ bias, float* __restrict__ out,
                               int Cin, int Hin, int Win, int Hout, int Wout,
                               int stride, int act){
  int p = blockIdx.x*blockDim.x + threadIdx.x;
  if (p >= Hout*Wout) return;
  int y = p / Wout, x = p - y*Wout;
  float acc = bias[0];
  for (int ci = 0; ci < Cin; ++ci){
    const float* ip = in + (size_t)ci*Hin*Win;
    const float* wp = w + ci*9;
#pragma unroll
    for (int ky = 0; ky < 3; ++ky){
      int iy = y*stride + ky - 1;
      if (iy < 0 || iy >= Hin) continue;
#pragma unroll
      for (int kx = 0; kx < 3; ++kx){
        int ix = x*stride + kx - 1;
        if (ix < 0 || ix >= Win) continue;
        acc += wp[ky*3 + kx] * ip[(size_t)iy*Win + ix];
      }
    }
  }
  if (act == 1) acc = acc > 0.f ? acc : 0.f;
  else if (act == 2) acc = sigm_(acc);
  out[p] = acc;
}

// ---------------- projection prep + warp ------------------------------------
__device__ void invert4x4_(const float* m, float* inv){
  float a[4][8];
  for (int i = 0; i < 4; ++i)
    for (int j = 0; j < 4; ++j){ a[i][j] = m[i*4+j]; a[i][j+4] = (i==j) ? 1.f : 0.f; }
  for (int col = 0; col < 4; ++col){
    int piv = col; float mx = fabsf(a[col][col]);
    for (int r = col+1; r < 4; ++r) if (fabsf(a[r][col]) > mx){ mx = fabsf(a[r][col]); piv = r; }
    if (piv != col) for (int j = 0; j < 8; ++j){ float t=a[col][j]; a[col][j]=a[piv][j]; a[piv][j]=t; }
    float id = 1.f / a[col][col];
    for (int j = 0; j < 8; ++j) a[col][j] *= id;
    for (int r = 0; r < 4; ++r){
      if (r == col) continue;
      float f = a[r][col];
      for (int j = 0; j < 8; ++j) a[r][j] -= f * a[col][j];
    }
  }
  for (int i = 0; i < 4; ++i) for (int j = 0; j < 4; ++j) inv[i*4+j] = a[i][j+4];
}
__global__ void prep_proj_k(const float* __restrict__ projs, float* __restrict__ rt, int V){
  if (threadIdx.x || blockIdx.x) return;
  float inv[16];
  invert4x4_(projs, inv);                   // ref = view 0
  for (int s = 1; s < V; ++s){
    const float* sp = projs + 16*s;
    float pm[16];
    for (int i = 0; i < 4; ++i)
      for (int j = 0; j < 4; ++j){
        float acc = 0.f;
        for (int k = 0; k < 4; ++k) acc += sp[i*4+k] * inv[k*4+j];
        pm[i*4+j] = acc;
      }
    float* o = rt + (s-1)*12;
    for (int i = 0; i < 3; ++i){
      for (int j = 0; j < 3; ++j) o[i*3+j] = pm[i*4+j];
      o[9+i] = pm[i*4+3];
    }
  }
}
__global__ void warp_diff_k(const float* __restrict__ src, const float* __restrict__ ref,
                            const float* __restrict__ rt, const float* __restrict__ dvals,
                            int dIdx, float* __restrict__ diff, int C, int H, int W){
  int p = blockIdx.x*blockDim.x + threadIdx.x;
  if (p >= H*W) return;
  float x = (float)(p % W), y = (float)(p / W);
  float d = dvals[dIdx];
  float px = (rt[0]*x + rt[1]*y + rt[2]) * d + rt[9];
  float py = (rt[3]*x + rt[4]*y + rt[5]) * d + rt[10];
  float pz = (rt[6]*x + rt[7]*y + rt[8]) * d + rt[11];
  if (fabsf(pz) < 1e-6f) pz = 1e-6f;
  float sx = px / pz, sy = py / pz;
  float x0f = floorf(sx), y0f = floorf(sy);
  float wx1 = sx - x0f, wx0 = 1.f - wx1, wy1 = sy - y0f, wy0 = 1.f - wy1;
  int x0 = (int)x0f, y0 = (int)y0f;
  float v00 = (x0   >= 0 && x0   < W && y0   >= 0 && y0   < H) ? wx0*wy0 : 0.f;
  float v10 = (x0+1 >= 0 && x0+1 < W && y0   >= 0 && y0   < H) ? wx1*wy0 : 0.f;
  float v01 = (x0   >= 0 && x0   < W && y0+1 >= 0 && y0+1 < H) ? wx0*wy1 : 0.f;
  float v11 = (x0+1 >= 0 && x0+1 < W && y0+1 >= 0 && y0+1 < H) ? wx1*wy1 : 0.f;
  int cx0 = min(max(x0,   0), W-1), cx1 = min(max(x0+1, 0), W-1);
  int cy0 = min(max(y0,   0), H-1), cy1 = min(max(y0+1, 0), H-1);
  for (int c = 0; c < C; ++c){
    const float* f = src + (size_t)c*H*W;
    float v = v00*f[(size_t)cy0*W+cx0] + v10*f[(size_t)cy0*W+cx1]
            + v01*f[(size_t)cy1*W+cx0] + v11*f[(size_t)cy1*W+cx1];
    float dl = v - ref[(size_t)c*H*W + p];
    diff[(size_t)c*H*W + p] = dl*dl;
  }
}

// ---------------- softmax depth regression + confidence ---------------------
__global__ void finalize_k(const float* __restrict__ costs, const float* __restrict__ dv,
                           float* __restrict__ depth, float* __restrict__ conf, int HW){
  int p = blockIdx.x*blockDim.x + threadIdx.x;
  if (p >= HW) return;
  float pr[ND];
  float mx = -3.4e38f;
  for (int d = 0; d < ND; ++d){ float v = costs[(size_t)d*HW + p]; pr[d] = v; mx = fmaxf(mx, v); }
  float sum = 0.f;
  for (int d = 0; d < ND; ++d){ float e = __expf(pr[d] - mx); pr[d] = e; sum += e; }
  float inv = 1.f / sum;
  float dep = 0.f, fidx = 0.f;
  for (int d = 0; d < ND; ++d){ float q = pr[d]*inv; pr[d] = q; dep += q*dv[d]; fidx += q*(float)d; }
  int di = (int)floorf(fidx);
  di = di < 0 ? 0 : (di > ND-1 ? ND-1 : di);
  float c4 = 0.f;
  for (int j = di-1; j <= di+2; ++j) if (j >= 0 && j < ND) c4 += pr[j];
  depth[p] = dep; conf[p] = c4;
}

// ===========================================================================
// Host orchestration
// ===========================================================================
static inline unsigned GRD(size_t n){ return (unsigned)((n + 255) / 256); }

extern "C" void kernel_launch(void* const* d_in, const int* in_sizes, int n_in,
                              void* d_out, int out_size, void* d_ws, size_t ws_size,
                              hipStream_t stream){
  (void)in_sizes; (void)n_in; (void)out_size; (void)ws_size;

  // ---- input index map (setup_inputs dict insertion order) -----------------
  const float* imgs  = (const float*)d_in[0];   // (1,3,3,384,512)
  const float* projs = (const float*)d_in[1];   // (1,3,4,4)
  const float* dvals = (const float*)d_in[2];   // (1,24)
  auto P = [&](int i)->const float*{ return (const float*)d_in[i]; };
  int fw[8], fb[8], fg[8], fbt[8];
  { int c = 3;
    for (int i = 0; i < 7; ++i){ fw[i]=c++; fb[i]=c++; fg[i]=c++; fbt[i]=c++; }
    fw[7]=31; fb[7]=32; }
  const int gw0=33, gb0=34, gg0=35, gbt0=36, gw1=37, gb1=38, gg1=39, gbt1=40, gw2=41, gb2=42;
  const int sw0=43, sb0=44, sw1=45, sb1=46, sw2=47, sb2=48;
  int lw[5], lb[5]; { int c = 49; for (int i = 0; i < 5; ++i){ lw[i]=c++; lb[i]=c++; } }
  const int cw=59, cb=60;

  // ---- workspace arena -----------------------------------------------------
  char* ws = (char*)d_ws; size_t cur = 0;
  auto alloc = [&](size_t nbytes)->void*{
    void* p = ws + cur; cur = (cur + nbytes + 255) & ~(size_t)255; return p; };
  auto allocF = [&](size_t nf)->float*{ return (float*)alloc(nf*sizeof(float)); };

  struct PW { _Float16* p; };
  auto packW = [&](int widx, int Cin, int Cout)->PW{
    int Kpad = ((Cin*9) + 31) & ~31, CoutPad = (Cout + 15) & ~15;
    _Float16* buf = (_Float16*)alloc((size_t)CoutPad*Kpad*sizeof(_Float16));
    int total = CoutPad*Kpad;
    pack_w_k<<<GRD(total), 256, 0, stream>>>(P(widx), buf, Cout, Cin, CoutPad, Kpad);
    return {buf};
  };

  const int fCin[8]  = {3,8,8,16,16,16,32,32};
  const int fCout[8] = {8,8,16,16,16,32,32,32};
  const int fStr[8]  = {1,1,2,1,1,2,1,1};
  PW wF[8]; for (int i = 0; i < 8; ++i) wF[i] = packW(fw[i], fCin[i], fCout[i]);
  PW wG0 = packW(gw0, 32, 16), wG1 = packW(gw1, 16, 4);
  PW wS0 = packW(sw0, 3, 8),   wS1 = packW(sw1, 8, 8);
  const int lCin[5]  = {32,16,16,32,32};
  const int lChid[5] = {16,16,16,16,8};
  PW wL[5]; for (int i = 0; i < 5; ++i) wL[i] = packW(lw[i], lCin[i]+lChid[i], 4*lChid[i]);

  // shared im2col buffer: max over all convs = L1 (Kpad=96, NpxPad=196608)
  _Float16* Bm = (_Float16*)alloc((size_t)96*196608*sizeof(_Float16));

  float* stats = allocF(8);
  float* rt    = allocF(24);
  float* ping  = allocF((size_t)8*H_IMG*W_IMG);
  float* pong  = allocF((size_t)8*H_IMG*W_IMG);
  float* feat[3]; for (int v = 0; v < 3; ++v) feat[v] = allocF((size_t)32*HWF);
  float* semA  = allocF((size_t)8*192*256);
  float* semB  = allocF((size_t)8*HWF);
  float* diffB = allocF((size_t)32*HWF);
  float* accB  = allocF((size_t)32*HWF);
  float* g0o   = allocF((size_t)16*HWF);
  float* g1o   = allocF((size_t)4*HWF);
  float* rwB   = allocF((size_t)HWF);
  const int sC[5] = {16,16,16,16,8};
  const int sH[5] = {HF, HF/2, HF/4, HF/2, HF};
  const int sW[5] = {WF, WF/2, WF/4, WF/2, WF};
  float *hS[5], *cS[5];
  for (int i = 0; i < 5; ++i){
    size_t n = (size_t)sC[i]*sH[i]*sW[i];
    hS[i] = allocF(n); cS[i] = allocF(n);
  }
  float* mp0  = allocF((size_t)16*(HF/2)*(WF/2));
  float* mp1  = allocF((size_t)16*(HF/4)*(WF/4));
  float* xc3  = allocF((size_t)32*(HF/2)*(WF/2));
  float* xc4  = allocF((size_t)32*HWF);
  float* catB = allocF((size_t)48*HWF);
  float* zB   = allocF((size_t)64*HWF);
  float* costs= allocF((size_t)ND*HWF);

  float* out = (float*)d_out;           // [depth | conf | semantic], each HWF

  // ---- launch helpers ------------------------------------------------------
  auto conv = [&](const float* in, PW w, const float* bias, float* o,
                  int Cin, int Cout, int Hin, int Win, int stride){
    int Hout = (Hin + stride - 1) / stride, Wout = (Win + stride - 1) / stride;
    int Kpad = ((Cin*9) + 31) & ~31;
    int Npx = Hout*Wout;
    int NpxPad = (Npx + 31) & ~31;
    size_t tot = (size_t)NpxPad*Kpad;
    im2col_k<<<GRD(tot), 256, 0, stream>>>(in, Bm, Cin, Hin, Win, Hout, Wout,
                                           stride, Kpad, NpxPad);
    int nOcT = (Cout + 15) >> 4, nPxT2 = NpxPad >> 5;
    int waves = nOcT*nPxT2, blocks = (waves + 7) >> 3;
    gemm_wmma_k<<<blocks, 256, 0, stream>>>(w.p, Bm, bias, o, Cout, Npx, Kpad,
                                            nOcT, nPxT2);
  };
  auto gnrelu = [&](float* x, int C, int HW, const float* g, const float* bt){
    gn_stats_k<<<4, 256, 0, stream>>>(x, stats, C, HW, 4);
    gn_apply_k<<<GRD((size_t)C*HW), 256, 0, stream>>>(x, stats, g, bt, C, HW, 4, 1);
  };
  auto cpy  = [&](float* d, const float* s, size_t n){ copy_k<<<GRD(n),256,0,stream>>>(d,s,n); };
  auto lstm = [&](const float* x, int Cx, int lvl){
    int H = sH[lvl], W = sW[lvl], HW = H*W, chid = sC[lvl];
    cpy(catB, x, (size_t)Cx*HW);
    cpy(catB + (size_t)Cx*HW, hS[lvl], (size_t)chid*HW);
    conv(catB, wL[lvl], P(lb[lvl]), zB, Cx + chid, 4*chid, H, W, 1);
    lstm_gates_k<<<GRD((size_t)chid*HW), 256, 0, stream>>>(zB, hS[lvl], cS[lvl], chid, HW);
  };

  // ---- projection prep -----------------------------------------------------
  prep_proj_k<<<1, 32, 0, stream>>>(projs, rt, 3);

  // ---- semnet (image 0) ----------------------------------------------------
  conv(imgs, wS0, P(sb0), semA, 3, 8, H_IMG, W_IMG, 2);
  act_k<<<GRD((size_t)8*192*256), 256, 0, stream>>>(semA, (size_t)8*192*256, 1);
  conv(semA, wS1, P(sb1), semB, 8, 8, 192, 256, 2);
  act_k<<<GRD((size_t)8*HWF), 256, 0, stream>>>(semB, (size_t)8*HWF, 1);
  conv3x3_1out_k<<<GRD(HWF), 256, 0, stream>>>(semB, P(sw2), P(sb2), out + 2*HWF,
                                               8, HF, WF, HF, WF, 1, 2);

  // ---- featnet x3 ----------------------------------------------------------
  for (int v = 0; v < 3; ++v){
    const float* x = imgs + (size_t)v*3*H_IMG*W_IMG;
    int Hc = H_IMG, Wc = W_IMG;
    float* a = ping; float* b = pong;
    const float* src = x;
    for (int L = 0; L < 7; ++L){
      int Ho = (Hc + fStr[L]-1)/fStr[L], Wo = (Wc + fStr[L]-1)/fStr[L];
      conv(src, wF[L], P(fb[L]), a, fCin[L], fCout[L], Hc, Wc, fStr[L]);
      gnrelu(a, fCout[L], Ho*Wo, P(fg[L]), P(fbt[L]));
      src = a; float* t = a; a = b; b = t;
      Hc = Ho; Wc = Wo;
    }
    conv(src, wF[7], P(fb[7]), feat[v], 32, 32, Hc, Wc, 1);   // 32 x 96 x 128
  }

  // ---- zero LSTM states ----------------------------------------------------
  for (int i = 0; i < 5; ++i){
    size_t n = (size_t)sC[i]*sH[i]*sW[i];
    zero_k<<<GRD(n), 256, 0, stream>>>(hS[i], n);
    zero_k<<<GRD(n), 256, 0, stream>>>(cS[i], n);
  }

  // ---- depth sweep ---------------------------------------------------------
  for (int d = 0; d < ND; ++d){
    for (int s = 0; s < 2; ++s){
      warp_diff_k<<<GRD(HWF), 256, 0, stream>>>(feat[s+1], feat[0], rt + s*12,
                                                dvals, d, diffB, 32, HF, WF);
      conv(diffB, wG0, P(gb0), g0o, 32, 16, HF, WF, 1);
      gnrelu(g0o, 16, HWF, P(gg0), P(gbt0));
      conv(g0o, wG1, P(gb1), g1o, 16, 4, HF, WF, 1);
      gnrelu(g1o, 4, HWF, P(gg1), P(gbt1));
      conv3x3_1out_k<<<GRD(HWF), 256, 0, stream>>>(g1o, P(gw2), P(gb2), rwB,
                                                   4, HF, WF, HF, WF, 1, 2);
      acc_term_k<<<GRD((size_t)32*HWF), 256, 0, stream>>>(accB, diffB, rwB, 32, HWF, s == 0);
    }
    neg_half_k<<<GRD((size_t)32*HWF), 256, 0, stream>>>(accB, (size_t)32*HWF); // -var

    lstm(accB, 32, 0);
    maxpool2_k<<<GRD((size_t)16*(HF/2)*(WF/2)), 256, 0, stream>>>(hS[0], mp0, 16, HF, WF);
    lstm(mp0, 16, 1);
    maxpool2_k<<<GRD((size_t)16*(HF/4)*(WF/4)), 256, 0, stream>>>(hS[1], mp1, 16, HF/2, WF/2);
    lstm(mp1, 16, 2);
    up2_k<<<GRD((size_t)16*(HF/2)*(WF/2)), 256, 0, stream>>>(hS[2], xc3, 16, HF/2, WF/2);
    cpy(xc3 + (size_t)16*(HF/2)*(WF/2), hS[1], (size_t)16*(HF/2)*(WF/2));
    lstm(xc3, 32, 3);
    up2_k<<<GRD((size_t)16*HWF), 256, 0, stream>>>(hS[3], xc4, 16, HF, WF);
    cpy(xc4 + (size_t)16*HWF, hS[0], (size_t)16*HWF);
    lstm(xc4, 32, 4);
    conv3x3_1out_k<<<GRD(HWF), 256, 0, stream>>>(hS[4], P(cw), P(cb),
                                                 costs + (size_t)d*HWF,
                                                 8, HF, WF, HF, WF, 1, 0);
  }

  // ---- softmax depth regression + windowed confidence ----------------------
  finalize_k<<<GRD(HWF), 256, 0, stream>>>(costs, dvals, out, out + HWF, HWF);
}
